// PunctDecoder_37941741093246
// MI455X (gfx1250) — matrix operand order repaired and checked by reading, not compiled, not run on hardware
//
#include <hip/hip_runtime.h>
#include <math.h>

#define B_   64
#define T_   2048
#define ENC_ 512
#define U_   512
#define VO_  32
#define TCH  128
#define NCH  (T_/TCH)   // 16 chunks per batch row

typedef unsigned short u16t;
typedef __attribute__((ext_vector_type(16))) __bf16 v16bf;
typedef __attribute__((ext_vector_type(8)))  float  v8f;

__device__ __forceinline__ u16t f2bf(float x){
  union{float f; unsigned u;} v; v.f=x;
  unsigned r = (v.u + 0x7FFFu + ((v.u>>16)&1u))>>16;   // round-to-nearest-even
  return (u16t)r;
}
__device__ __forceinline__ float sigm(float x){ return 1.0f/(1.0f+__expf(-x)); }

// Async copy of one 16KB B-slab (global -> LDS), 16B per lane per op.
// GVS mode: global = SADDR + VADDR + inst_offset ; LDS = VDST + inst_offset.
__device__ __forceinline__ void async_copy_slab(unsigned lds_dst, unsigned g_off,
                                                const u16t* gbase){
  asm volatile(
    "global_load_async_to_lds_b128 %0, %1, %2 offset:0\n\t"
    "global_load_async_to_lds_b128 %0, %1, %2 offset:4096\n\t"
    "global_load_async_to_lds_b128 %0, %1, %2 offset:8192\n\t"
    "global_load_async_to_lds_b128 %0, %1, %2 offset:12288"
    :: "v"(lds_dst), "v"(g_off), "s"(gbase)
    : "memory");
}
__device__ __forceinline__ void wait_async0(){
  asm volatile("s_wait_asynccnt 0x0" ::: "memory");
}

// ---------------------------------------------------------------------------
// W1 (512x512 f32, row-major [k][u]) -> bf16, WMMA B-fragment tiled layout.
// Tile = (kb: 32 K's) x (ut: 16 N's), tile id = ut*16+kb, 512 u16t per tile.
// Within a tile the two 16-byte fragment halves are stored separately so LDS
// b128 reads have a 16B lane stride (bank-conflict-free):
//   idx = tile*512 + h*256 + lane*8 + jj      (h = j>>3, jj = j&7)
// Lane n (0-15): N=n, K=j (j<8) / 8+j (j>=8); lanes 16-31: N=lane-16, K=8+j (j<8) / 16+j (j>=8).
// ---------------------------------------------------------------------------
__global__ void k_prep_w1(const float* __restrict__ W1, u16t* __restrict__ W1t){
  int o = blockIdx.x*blockDim.x + threadIdx.x;   // 65536 threads
  for (int e=0;e<4;++e){
    int idx  = o + e*65536;                      // 0..262143
    int jj   = idx & 7;
    int lane = (idx>>3)&31;
    int h    = (idx>>8)&1;
    int tile = idx>>9;                           // 0..511
    int kb   = tile & 15;
    int ut   = tile >> 4;
    int j    = h*8 + jj;
    int k    = kb*32 + (lane>>4)*8 + h*16 + jj;
    int u    = ut*16 + (lane&15);
    W1t[idx] = f2bf(W1[k*U_ + u]);
  }
}

// hidden = s_t_m @ W2 + b2   [64,512]
__global__ void k_hidden(const float* __restrict__ s_t_m, const float* __restrict__ W2,
                         const float* __restrict__ b2, float* __restrict__ hidden){
  __shared__ float x[U_];
  int b = blockIdx.x, tid = threadIdx.x;
  x[tid] = s_t_m[b*U_+tid]; x[tid+256] = s_t_m[b*U_+tid+256];
  __syncthreads();
  for (int uu=0; uu<2; ++uu){
    int u = tid + uu*256;
    float acc = b2[u];
    for (int k=0;k<U_;++k) acc += x[k]*W2[k*U_+u];
    hidden[b*U_+u] = acc;
  }
}

// GRU single step (reset_after=true, [z,r,h] blocks); writes s_t to output + ws
__global__ void k_gru(const float* __restrict__ h_t, const float* __restrict__ h0g,
                      const float* __restrict__ Wx, const float* __restrict__ Wh,
                      const float* __restrict__ bi, const float* __restrict__ br,
                      float* __restrict__ st_out, float* __restrict__ st_ws){
  __shared__ float x[ENC_];
  __shared__ float h0[U_];
  int b=blockIdx.x, tid=threadIdx.x;
  x[tid]=h_t[b*ENC_+tid];   x[tid+256]=h_t[b*ENC_+tid+256];
  h0[tid]=h0g[b*U_+tid];    h0[tid+256]=h0g[b*U_+tid+256];
  __syncthreads();
  for (int uu=0; uu<2; ++uu){
    int u=tid+uu*256;
    float xz=bi[u], xr=bi[U_+u], xh=bi[2*U_+u];
    float hz=br[u], hr=br[U_+u], hh=br[2*U_+u];
    for (int k=0;k<ENC_;++k){
      float xv=x[k];  const float* wr = Wx+(size_t)k*3*U_;
      xz+=xv*wr[u]; xr+=xv*wr[U_+u]; xh+=xv*wr[2*U_+u];
      float hv=h0[k]; const float* hw = Wh+(size_t)k*3*U_;
      hz+=hv*hw[u]; hr+=hv*hw[U_+u]; hh+=hv*hw[2*U_+u];
    }
    float z=sigm(xz+hz), r=sigm(xr+hr);
    float hc=tanhf(xh + r*hh);
    float st = z*h0[u] + (1.0f-z)*hc;
    st_out[b*U_+u]=st; st_ws[b*U_+u]=st;
  }
}

// ---------------------------------------------------------------------------
// Attention main kernel: one WG per (b, 128-row t-chunk). 8 waves x 16 rows.
// B slabs (16KB per u-tile, shared by all 8 waves) are double-buffered in LDS
// via async global->LDS copies; A fragments hoisted in 128 VGPRs per wave;
// B fragments software-pipelined one K-step ahead in registers.
// ---------------------------------------------------------------------------
__global__ void __launch_bounds__(256) k_attn(
    const float* __restrict__ H, const u16t* __restrict__ W1t,
    const float* __restrict__ hidden, const float* __restrict__ W1b,
    const float* __restrict__ Vw, const float* __restrict__ Vb,
    float* __restrict__ chunk_m, float* __restrict__ chunk_s,
    float* __restrict__ chunk_ctx)
{
  __shared__ u16t Bs[2][8192];      // 2 x 16KB double buffer
  __shared__ float scores[TCH];
  __shared__ float wts[TCH];
  __shared__ float redbuf[TCH];

  int b   = blockIdx.x / NCH;
  int ch  = blockIdx.x % NCH;
  int t0  = ch*TCH;
  int tid = threadIdx.x;
  int wave = tid>>5, lane = tid&31;
  int m  = lane&15, kh = lane>>4;
  int row0 = wave*16;

  const float* Hb    = H + ((size_t)b*T_ + t0)*ENC_;
  const float* myrow = Hb + (size_t)(row0+m)*ENC_;

  // LDS byte offsets of the two slab buffers (low 32 bits of flat shared addr
  // == LDS offset on gfx1250).
  unsigned ldsA = (unsigned)(uintptr_t)(&Bs[0][0]) + (unsigned)tid*16u;
  unsigned ldsB = (unsigned)(uintptr_t)(&Bs[1][0]) + (unsigned)tid*16u;

  // Kick off slab 0 while we build the A fragments.
  async_copy_slab(ldsA, (unsigned)tid*16u, W1t);

  // Hoist all A fragments for K=0..511 (16 x v16bf = 128 VGPRs).
  // A 16x32 bf16 layout: lane<16: j0..7 -> K=k0..k0+7, j8..15 -> K=k0+16..+23
  //                      lane>=16: +8 on both K ranges.
  v16bf A[16];
#pragma unroll
  for (int kb=0; kb<16; ++kb){
    const float* p = myrow + kb*32 + kh*8;
    float4 a0 = *(const float4*)(p);
    float4 a1 = *(const float4*)(p+4);
    float4 b0 = *(const float4*)(p+16);
    float4 b1 = *(const float4*)(p+20);
    union { v16bf v; u16t s[16]; } Ua;
    Ua.s[0]=f2bf(a0.x);  Ua.s[1]=f2bf(a0.y);  Ua.s[2]=f2bf(a0.z);  Ua.s[3]=f2bf(a0.w);
    Ua.s[4]=f2bf(a1.x);  Ua.s[5]=f2bf(a1.y);  Ua.s[6]=f2bf(a1.z);  Ua.s[7]=f2bf(a1.w);
    Ua.s[8]=f2bf(b0.x);  Ua.s[9]=f2bf(b0.y);  Ua.s[10]=f2bf(b0.z); Ua.s[11]=f2bf(b0.w);
    Ua.s[12]=f2bf(b1.x); Ua.s[13]=f2bf(b1.y); Ua.s[14]=f2bf(b1.z); Ua.s[15]=f2bf(b1.w);
    A[kb]=Ua.v;
  }

  float sp[8];
#pragma unroll
  for (int r=0;r<8;++r) sp[r]=0.f;
  float vb = Vb[0];

  wait_async0();
  __syncthreads();           // slab 0 resident

  union Bfrag { v16bf v; uint4 q[2]; };

  for (int ut=0; ut<32; ++ut){
    // Prefetch next slab into the other buffer (it was fully consumed in the
    // previous iteration; the trailing barrier made that globally visible).
    if (ut < 31){
      unsigned dst = ((ut&1)^1) ? ldsB : ldsA;
      async_copy_slab(dst, (unsigned)(ut+1)*16384u + (unsigned)tid*16u, W1t);
    }

    const u16t* curbuf = &Bs[ut&1][0];
    v8f c = {};

    // Software-pipeline B fragments one K-step ahead (fully unrolled, so the
    // cur=nxt copies are renamed away and WMMA never waits on the just-issued
    // ds loads).
    Bfrag cur, nxt;
    cur.q[0] = *(const uint4*)(curbuf + lane*8);        // kb=0, j=0..7
    cur.q[1] = *(const uint4*)(curbuf + 256 + lane*8);  // kb=0, j=8..15
    nxt = cur;
#pragma unroll
    for (int kb=0; kb<16; ++kb){
      if (kb < 15){
        nxt.q[0] = *(const uint4*)(curbuf + (kb+1)*512 + lane*8);
        nxt.q[1] = *(const uint4*)(curbuf + (kb+1)*512 + 256 + lane*8);
      }
      c = __builtin_amdgcn_wmma_f32_16x16x32_bf16(false, A[kb], false, cur.v,
                                                  (short)0, c, false, false);
      cur = nxt;
    }

    // C layout: VGPR r -> row (row0 + r [+8 if lane>=16]), column = u0+(lane&15)
    int n = ut*16 + (lane&15);
    float hb = W1b[n] + hidden[b*U_+n];
    float vw = Vw[n];
#pragma unroll
    for (int r=0;r<8;++r) sp[r] += tanhf(c[r]+hb)*vw;

    wait_async0();
    __syncthreads();         // next slab resident; this slab free to overwrite
  }

  // reduce across the 16 columns held by each lane half
#pragma unroll
  for (int r=0;r<8;++r){
    sp[r] += __shfl_xor(sp[r], 1, 32);
    sp[r] += __shfl_xor(sp[r], 2, 32);
    sp[r] += __shfl_xor(sp[r], 4, 32);
    sp[r] += __shfl_xor(sp[r], 8, 32);
  }
  if ((lane&15)==0){
    int rb = row0 + (kh ? 8 : 0);
#pragma unroll
    for (int r=0;r<8;++r) scores[rb+r] = sp[r] + vb;
  }
  __syncthreads();

  // chunk max
  if (tid<TCH) redbuf[tid]=scores[tid];
  __syncthreads();
  for (int s=TCH/2; s>0; s>>=1){
    if (tid<s) redbuf[tid]=fmaxf(redbuf[tid], redbuf[tid+s]);
    __syncthreads();
  }
  float mC = redbuf[0];
  __syncthreads();
  float w=0.f;
  if (tid<TCH){ w=__expf(scores[tid]-mC); wts[tid]=w; redbuf[tid]=w; }
  __syncthreads();
  for (int s=TCH/2; s>0; s>>=1){
    if (tid<s) redbuf[tid]+=redbuf[tid+s];
    __syncthreads();
  }
  float sC = redbuf[0];
  if (tid==0){ chunk_m[b*NCH+ch]=mC; chunk_s[b*NCH+ch]=sC; }

  // partial context: Σ_t exp(s_t - mC) * H[t][u]  (H tile is L2-hot)
  float acc0=0.f, acc1=0.f;
  int u = tid;
  for (int t=0;t<TCH;++t){
    float wv = wts[t];
    const float* hr = Hb + (size_t)t*ENC_;
    acc0 += wv*hr[u];
    acc1 += wv*hr[u+256];
  }
  float* cc = chunk_ctx + (size_t)(b*NCH+ch)*ENC_;
  cc[u]=acc0; cc[u+256]=acc1;
}

// merge chunk partials -> context, then proj = context @ Wfa + bfa
__global__ void k_combine(const float* __restrict__ chunk_m, const float* __restrict__ chunk_s,
                          const float* __restrict__ chunk_ctx,
                          const float* __restrict__ Wfa, const float* __restrict__ bfa,
                          float* __restrict__ proj){
  __shared__ float ctx[ENC_];
  __shared__ float fac[NCH];
  __shared__ float sinv;
  int b=blockIdx.x, tid=threadIdx.x;
  if (tid==0){
    float M=-3.0e38f;
    for(int c=0;c<NCH;++c) M=fmaxf(M, chunk_m[b*NCH+c]);
    float tot=0.f;
    for(int c=0;c<NCH;++c){ float f=__expf(chunk_m[b*NCH+c]-M); fac[c]=f; tot+=chunk_s[b*NCH+c]*f; }
    sinv = 1.0f/tot;
  }
  __syncthreads();
  for (int uu=0; uu<2; ++uu){
    int u = tid+uu*256;
    float acc=0.f;
    for (int c=0;c<NCH;++c) acc += chunk_ctx[(size_t)(b*NCH+c)*ENC_+u]*fac[c];
    ctx[u]=acc*sinv;
  }
  __syncthreads();
  for (int uu=0; uu<2; ++uu){
    int u = tid+uu*256;
    float acc=bfa[u];
    for(int k=0;k<ENC_;++k) acc += ctx[k]*Wfa[k*U_+u];
    proj[b*U_+u]=acc;
  }
}

// f_t = proj * sigmoid(proj@Wff + bff + s_t@Wfh + bfh) + s_t
__global__ void k_fusion(const float* __restrict__ proj, const float* __restrict__ st,
                         const float* __restrict__ Wff, const float* __restrict__ bff,
                         const float* __restrict__ Wfh, const float* __restrict__ bfh,
                         float* __restrict__ ft){
  __shared__ float p[U_]; __shared__ float s[U_];
  int b=blockIdx.x, tid=threadIdx.x;
  p[tid]=proj[b*U_+tid]; p[tid+256]=proj[b*U_+tid+256];
  s[tid]=st[b*U_+tid];   s[tid+256]=st[b*U_+tid+256];
  __syncthreads();
  for (int uu=0; uu<2; ++uu){
    int u=tid+uu*256;
    float a=bff[u]+bfh[u];
    for(int k=0;k<U_;++k) a += p[k]*Wff[k*U_+u] + s[k]*Wfh[k*U_+u];
    float fw = sigm(a);
    ft[b*U_+u] = p[u]*fw + s[u];
  }
}

// y = f_t @ Wy + by   [64,32]
__global__ void k_y(const float* __restrict__ ft, const float* __restrict__ Wy,
                    const float* __restrict__ by, float* __restrict__ y){
  __shared__ float f[U_];
  int b=blockIdx.x, tid=threadIdx.x;     // 128 threads
  for (int i=tid;i<U_;i+=128) f[i]=ft[b*U_+i];
  __syncthreads();
  if (tid<VO_){
    float acc=by[tid];
    for(int k=0;k<U_;++k) acc += f[k]*Wy[k*VO_+tid];
    y[b*VO_+tid]=acc;
  }
}

extern "C" void kernel_launch(void* const* d_in, const int* in_sizes, int n_in,
                              void* d_out, int out_size, void* d_ws, size_t ws_size,
                              hipStream_t stream){
  (void)in_sizes; (void)n_in; (void)out_size; (void)ws_size;
  const float* h_t  = (const float*)d_in[0];
  const float* s_t_m= (const float*)d_in[1];
  const float* H    = (const float*)d_in[2];
  const float* h0   = (const float*)d_in[3];
  const float* W1   = (const float*)d_in[4];
  const float* W1b  = (const float*)d_in[5];
  const float* W2   = (const float*)d_in[6];
  const float* W2b  = (const float*)d_in[7];
  const float* Vw   = (const float*)d_in[8];
  const float* Vb   = (const float*)d_in[9];
  const float* Wx   = (const float*)d_in[10];
  const float* Wh   = (const float*)d_in[11];
  const float* bi   = (const float*)d_in[12];
  const float* br   = (const float*)d_in[13];
  const float* Wfa  = (const float*)d_in[14];
  const float* bfa  = (const float*)d_in[15];
  const float* Wff  = (const float*)d_in[16];
  const float* bff  = (const float*)d_in[17];
  const float* Wfh  = (const float*)d_in[18];
  const float* bfh  = (const float*)d_in[19];
  const float* Wy   = (const float*)d_in[20];
  const float* by   = (const float*)d_in[21];

  char* ws = (char*)d_ws;
  u16t*  W1t       = (u16t*) (ws + 0);          // 512KB
  float* hidden    = (float*)(ws + 524288);     // 128KB
  float* st_ws     = (float*)(ws + 655360);     // 128KB
  float* chunk_m   = (float*)(ws + 786432);     // 4KB
  float* chunk_s   = (float*)(ws + 790528);     // 4KB
  float* chunk_ctx = (float*)(ws + 794624);     // 2MB
  float* proj      = (float*)(ws + 2891776);    // 128KB
  float* ft        = (float*)(ws + 3022848);    // 128KB

  float* y_out  = (float*)d_out;                // [64,32]
  float* st_out = (float*)d_out + B_*VO_;       // [64,512]

  k_prep_w1<<<256, 256, 0, stream>>>(W1, W1t);
  k_hidden <<<B_,  256, 0, stream>>>(s_t_m, W2, W2b, hidden);
  k_gru    <<<B_,  256, 0, stream>>>(h_t, h0, Wx, Wh, bi, br, st_out, st_ws);
  k_attn   <<<B_*NCH, 256, 0, stream>>>(H, W1t, hidden, W1b, Vw, Vb,
                                        chunk_m, chunk_s, chunk_ctx);
  k_combine<<<B_,  256, 0, stream>>>(chunk_m, chunk_s, chunk_ctx, Wfa, bfa, proj);
  k_fusion <<<B_,  256, 0, stream>>>(proj, st_ws, Wff, bff, Wfh, bfh, ft);
  k_y      <<<B_,  128, 0, stream>>>(ft, Wy, by, y_out);
}
